// MultiheadSelfAttention_66460323939032
// MI455X (gfx1250) — matrix operand order repaired and checked by reading, not compiled, not run on hardware
//
#include <hip/hip_runtime.h>
#include <hip/hip_bf16.h>

// ---------------------------------------------------------------------------
// MI455X / gfx1250 fused causal multi-head self-attention
//   B=2, S=2048, D_MODEL=1024, H=16, D_K=64, fp32 in/out, bf16 WMMA compute.
//   GEMMs: 128x64 per block, TDM-staged weight panel in LDS (double buffered),
//   16x64 per wave (A-fragment reuse x4). Attention: per-wave flash tiles.
// ---------------------------------------------------------------------------

typedef __bf16 bf16_t;
typedef __attribute__((ext_vector_type(16))) __bf16 v16bf;
typedef __attribute__((ext_vector_type(8)))  __bf16 v8bf;
typedef __attribute__((ext_vector_type(8)))  float  v8f;
typedef __attribute__((ext_vector_type(4)))  unsigned int u32x4;
typedef __attribute__((ext_vector_type(4)))  int i32x4;
typedef __attribute__((ext_vector_type(8)))  int i32x8;

#define DM   1024
#define SEQ  2048
#define NH   16
#define DK   64
#define NB   2

__device__ __forceinline__ bf16_t f2bf(float f) {
    unsigned u = __builtin_bit_cast(unsigned, f);
    u += 0x7FFFu + ((u >> 16) & 1u);            // round-to-nearest-even
    unsigned short h = (unsigned short)(u >> 16);
    return __builtin_bit_cast(bf16_t, h);
}

// A-fragment (16x32 bf16, MxK): lane<16 holds row r=lane, K [k0..k0+15];
// lane>=16 holds row r=lane-16, K [k0+16..k0+31].
__device__ __forceinline__ v16bf load_a_frag(const bf16_t* base, int ld, int k0, int lane) {
    const int half = lane >> 4, r = lane & 15;
    const bf16_t* p = base + (size_t)r * ld + k0 + half * 16;
    v8bf lo = *(const v8bf*)p;
    v8bf hi = *(const v8bf*)(p + 8);
    return __builtin_shufflevector(lo, hi, 0,1,2,3,4,5,6,7,8,9,10,11,12,13,14,15);
}

// B-fragment (32x16 bf16, KxN), source stored column-contiguous ([N][K] row-major):
// lane holds column n=lane&15, contiguous K range (lane>=16 ? k0+16 : k0)..+15.
__device__ __forceinline__ v16bf load_b_frag(const bf16_t* base, int ld, int k0, int lane) {
    const int n = lane & 15, kh = (lane >> 4) * 16;
    return *(const v16bf*)(base + (size_t)n * ld + k0 + kh);
}

// B-fragment from a [64][32] bf16 LDS panel; t selects the 16-column sub-tile.
__device__ __forceinline__ v16bf load_b_frag_lds(const bf16_t* base, int t, int lane) {
    const int n = t * 16 + (lane & 15), kh = (lane >> 4) * 16;
    return *(const v16bf*)(base + n * 32 + kh);
}

__device__ __forceinline__ v8f wmma_bf16(v16bf a, v16bf b, v8f c) {
    return __builtin_amdgcn_wmma_f32_16x16x32_bf16(false, a, false, b,
                                                   (short)0, c, false, false);
}

// LDS byte offset of a __shared__ object (addrspace(3) ptrtoint).
__device__ __forceinline__ unsigned lds_offset(const void* p) {
    typedef __attribute__((address_space(3))) const void* lds_vp;
    return (unsigned)(unsigned long long)(lds_vp)p;
}

__device__ __forceinline__ int rfl(unsigned v) {
    return __builtin_amdgcn_readfirstlane((int)v);
}

// ---------------------------------------------------------------------------
// Tensor Data Mover: async load of a 2D bf16 tile (32 contiguous K elements x
// `rows` rows, row stride `stride` elements) from global into LDS.
// D# built per ISA 8.3/8.4: group0 = {count=1, lds_addr, global_addr, type=2},
// group1 = {mask=0, data_size=2B, tensor dims (oversized), tile 32 x rows,
//           dim0 stride}. Tracked with TENSORcnt.
// ---------------------------------------------------------------------------
__device__ __forceinline__ void tdm_load_panel(unsigned lds_off, const bf16_t* gptr,
                                               int rows, int stride_elems) {
    const unsigned long long ga = (unsigned long long)gptr;
    const unsigned td = 1u << 20;               // oversized tensor dims (no OOB clip)
    u32x4 g0;
    g0.x = (unsigned)rfl(1u);                                   // count=1
    g0.y = (unsigned)rfl(lds_off);                              // lds_addr
    g0.z = (unsigned)rfl((unsigned)ga);                         // global_addr[31:0]
    g0.w = (unsigned)rfl((unsigned)(ga >> 32) | 0x80000000u);   // addr[56:32]|type=2
    i32x8 g1;
    g1[0] = rfl(0x00010000u);                                   // data_size=2B
    g1[1] = rfl((td & 0xFFFFu) << 16);                          // tensor_dim0 lo16
    g1[2] = rfl((td >> 16) | ((td & 0xFFFFu) << 16));           // dim0 hi / dim1 lo
    g1[3] = rfl((td >> 16) | (32u << 16));                      // dim1 hi / tile_dim0=32
    g1[4] = rfl((unsigned)rows);                                // tile_dim1, tile_dim2=0
    g1[5] = rfl((unsigned)stride_elems);                        // tensor_dim0_stride
    g1[6] = rfl(0u);
    g1[7] = rfl(0u);
    i32x4 z = {0, 0, 0, 0};
#if __clang_major__ >= 23
    i32x8 z8 = {0, 0, 0, 0, 0, 0, 0, 0};
    __builtin_amdgcn_tensor_load_to_lds(g0, g1, z, z, z8, 0);
#else
    __builtin_amdgcn_tensor_load_to_lds(g0, g1, z, z, 0);
#endif
}

// ---------------------------------------------------------------------------
// Kernel 1: fp32 -> bf16 conversion
// ---------------------------------------------------------------------------
__global__ __launch_bounds__(256) void cvt_f32_bf16(const float* __restrict__ src,
                                                    bf16_t* __restrict__ dst, int n) {
    int i = blockIdx.x * 256 + threadIdx.x;
    if (i < n) dst[i] = f2bf(src[i]);
}

// ---------------------------------------------------------------------------
// Kernel 2: QKV projection (x @ W_qkv^T) fused with RoPE.
//   Block = 128 rows x 64 cols (8 waves x 16x64). W panel (64x32 bf16 per
//   k-step) TDM-staged into double-buffered LDS and shared by all waves.
//   Q,K -> [B,H,S,64] bf16 (RoPE applied); V -> transposed [B,H,64,S] bf16.
// ---------------------------------------------------------------------------
__global__ __launch_bounds__(256) void qkv_rope_kernel(
        const bf16_t* __restrict__ Xb, const bf16_t* __restrict__ Wb,
        const int* __restrict__ tokpos,
        bf16_t* __restrict__ Q, bf16_t* __restrict__ K, bf16_t* __restrict__ Vt) {
    __shared__ __align__(32) bf16_t bpanel[2][64 * 32];

    const int wave = threadIdx.x >> 5;
    const int lane = threadIdx.x & 31;
    const int mblk = blockIdx.x / 48, nblk = blockIdx.x % 48;   // 32 x 48 blocks
    const int m0 = mblk * 128 + wave * 16;
    const int n0 = nblk * 64;
    const bf16_t* Arow = Xb + (size_t)m0 * DM;
    const bf16_t* Wrow = Wb + (size_t)n0 * DM;

    const unsigned lds0 = lds_offset(&bpanel[0][0]);
    const unsigned lds1 = lds_offset(&bpanel[1][0]);
    const int NK = DM / 32;

    if (wave == 0) tdm_load_panel(lds0, Wrow, 64, DM);          // prefetch k=0

    v8f acc[4] = {};
    int cur = 0;
    for (int k = 0; k < NK; ++k) {
        if (wave == 0) {
            if (k + 1 < NK) {
                tdm_load_panel(cur ? lds0 : lds1, Wrow + (k + 1) * 32, 64, DM);
                __builtin_amdgcn_s_wait_tensorcnt(1);           // current panel ready
            } else {
                __builtin_amdgcn_s_wait_tensorcnt(0);
            }
        }
        __syncthreads();                                        // publish panel
        const v16bf a = load_a_frag(Arow, DM, k * 32, lane);
        const bf16_t* pan = &bpanel[cur][0];
        #pragma unroll
        for (int t = 0; t < 4; ++t)
            acc[t] = wmma_bf16(a, load_b_frag_lds(pan, t, lane), acc[t]);
        __syncthreads();                                        // panel free for reuse
        cur ^= 1;
    }

    const int half = lane >> 4, col = lane & 15;
    const int sec = n0 / DM;            // 0=Q 1=K 2=V (uniform: 64 | 1024)
    const int h   = (n0 % DM) / DK;     // head (uniform: n0 is 64-aligned)

    if (sec < 2) {
        bf16_t* dst = (sec == 0) ? Q : K;
        #pragma unroll
        for (int t = 0; t < 4; ++t) {
            const int d = t * 16 + col;
            const float freq = __powf(10000.0f, -(float)(d & ~1) / (float)DK);
            #pragma unroll
            for (int j = 0; j < 8; ++j) {
                const int m = m0 + j + half * 8;
                const int bb = m >> 11, s = m & (SEQ - 1);
                const float pos = (float)tokpos[s];
                float sn, cs;
                __sincosf(pos * freq, &sn, &cs);
                const float v  = acc[t][j];
                const float pv = __shfl_xor(v, 1, 32);          // even<->odd partner
                const float r  = ((d & 1) == 0) ? (v * cs - pv * sn)
                                                : (pv * sn + v * cs);
                dst[(((size_t)bb * NH + h) * SEQ + s) * DK + d] = f2bf(r);
            }
        }
    } else {
        #pragma unroll
        for (int t = 0; t < 4; ++t) {
            const int d = t * 16 + col;
            #pragma unroll
            for (int j = 0; j < 8; ++j) {
                const int m = m0 + j + half * 8;
                const int bb = m >> 11, s = m & (SEQ - 1);
                Vt[(((size_t)bb * NH + h) * DK + d) * SEQ + s] = f2bf(acc[t][j]);
            }
        }
    }
}

// ---------------------------------------------------------------------------
// Kernel 3: causal flash attention. One wave per (b,h,16-row q-tile).
//   scores = Q K^T / 8 (two 16x16 tiles per 32-key block, 2 WMMAs each),
//   online softmax in fp32, P converted to bf16 A-layout via per-wave LDS,
//   out += P V via 4 WMMAs (V pre-transposed -> contiguous B fragments).
// ---------------------------------------------------------------------------
__global__ __launch_bounds__(256) void flash_attn_kernel(
        const bf16_t* __restrict__ Q, const bf16_t* __restrict__ K,
        const bf16_t* __restrict__ Vt, bf16_t* __restrict__ AO) {
    __shared__ __align__(32) bf16_t pbuf[8][16 * 32];

    const int wave = threadIdx.x >> 5, lane = threadIdx.x & 31;
    const int half = lane >> 4, col = lane & 15;
    const int wt = blockIdx.x * 8 + wave;        // 32 * 128 tiles
    const int qt = wt & 127;                     // q-tile within sequence
    const int bh = wt >> 7;                      // fused (b,h)

    const bf16_t* Qbh = Q  + (size_t)bh * SEQ * DK;
    const bf16_t* Kbh = K  + (size_t)bh * SEQ * DK;
    const bf16_t* Vbh = Vt + (size_t)bh * DK * SEQ;

    const v16bf qa0 = load_a_frag(Qbh + (size_t)qt * 16 * DK, DK,  0, lane);
    const v16bf qa1 = load_a_frag(Qbh + (size_t)qt * 16 * DK, DK, 32, lane);

    v8f acc0 = {}, acc1 = {}, acc2 = {}, acc3 = {};
    float mrow[8], lrow[8];
    #pragma unroll
    for (int j = 0; j < 8; ++j) { mrow[j] = -3.0e38f; lrow[j] = 0.0f; }

    bf16_t* myp = &pbuf[wave][0];
    const int nkb = (qt * 16 + 47) >> 5;         // causal: key blocks of 32
    for (int kb = 0; kb < nkb; ++kb) {
        // ---- scores: two 16x16 tiles, contraction over d=64 --------------
        v8f sc0 = {}, sc1 = {};
        {
            const bf16_t* k0p = Kbh + (size_t)(kb * 32) * DK;
            sc0 = wmma_bf16(qa0, load_b_frag(k0p, DK,  0, lane), sc0);
            sc0 = wmma_bf16(qa1, load_b_frag(k0p, DK, 32, lane), sc0);
            const bf16_t* k1p = k0p + 16 * DK;
            sc1 = wmma_bf16(qa0, load_b_frag(k1p, DK,  0, lane), sc1);
            sc1 = wmma_bf16(qa1, load_b_frag(k1p, DK, 32, lane), sc1);
        }
        // ---- online softmax ---------------------------------------------
        const int key0 = kb * 32 + col;
        const int key1 = key0 + 16;
        #pragma unroll
        for (int j = 0; j < 8; ++j) {
            const int qrow = qt * 16 + j + half * 8;
            float s0 = sc0[j] * 0.125f; if (key0 > qrow) s0 = -3.0e38f;
            float s1 = sc1[j] * 0.125f; if (key1 > qrow) s1 = -3.0e38f;
            float mx = fmaxf(s0, s1);
            #pragma unroll
            for (int off = 1; off < 16; off <<= 1)
                mx = fmaxf(mx, __shfl_xor(mx, off, 32));
            const float mnew = fmaxf(mrow[j], mx);
            const float p0 = __expf(s0 - mnew);
            const float p1 = __expf(s1 - mnew);
            float rs = p0 + p1;
            #pragma unroll
            for (int off = 1; off < 16; off <<= 1)
                rs += __shfl_xor(rs, off, 32);
            const float corr = __expf(mrow[j] - mnew);
            lrow[j] = lrow[j] * corr + rs;
            mrow[j] = mnew;
            acc0[j] *= corr; acc1[j] *= corr; acc2[j] *= corr; acc3[j] *= corr;
            // stash P tile row-major [16][32] in this wave's LDS slice
            const int r = j + half * 8;
            myp[r * 32 + col]      = f2bf(p0);
            myp[r * 32 + 16 + col] = f2bf(p1);
        }
        __asm volatile("s_wait_dscnt 0x0" ::: "memory");
        const v16bf pa = load_a_frag(myp, 32, 0, lane);   // ds_load, A layout
        // ---- out += P V  (V^T gives contiguous B fragments) --------------
        const bf16_t* vb = Vbh + kb * 32;
        acc0 = wmma_bf16(pa, load_b_frag(vb + (size_t)0 * 16 * SEQ, SEQ, 0, lane), acc0);
        acc1 = wmma_bf16(pa, load_b_frag(vb + (size_t)1 * 16 * SEQ, SEQ, 0, lane), acc1);
        acc2 = wmma_bf16(pa, load_b_frag(vb + (size_t)2 * 16 * SEQ, SEQ, 0, lane), acc2);
        acc3 = wmma_bf16(pa, load_b_frag(vb + (size_t)3 * 16 * SEQ, SEQ, 0, lane), acc3);
    }

    // ---- epilogue: normalize, store bf16 head-concat [B,S,1024] ----------
    const int bb = bh >> 4, h = bh & 15;
    #pragma unroll
    for (int j = 0; j < 8; ++j) {
        const float inv = 1.0f / lrow[j];
        const int s = qt * 16 + j + half * 8;
        const size_t rowb = ((size_t)bb * SEQ + s) * DM + h * DK;
        AO[rowb + 0 * 16 + col] = f2bf(acc0[j] * inv);
        AO[rowb + 1 * 16 + col] = f2bf(acc1[j] * inv);
        AO[rowb + 2 * 16 + col] = f2bf(acc2[j] * inv);
        AO[rowb + 3 * 16 + col] = f2bf(acc3[j] * inv);
    }
}

// ---------------------------------------------------------------------------
// Kernel 4: output projection  out = AO @ W_out^T  (fp32 result).
//   Same 128x64-per-block TDM-staged structure as the QKV GEMM.
// ---------------------------------------------------------------------------
__global__ __launch_bounds__(256) void out_proj_kernel(
        const bf16_t* __restrict__ AO, const bf16_t* __restrict__ Wb,
        float* __restrict__ out) {
    __shared__ __align__(32) bf16_t bpanel[2][64 * 32];

    const int wave = threadIdx.x >> 5, lane = threadIdx.x & 31;
    const int mblk = blockIdx.x / 16, nblk = blockIdx.x % 16;   // 32 x 16 blocks
    const int m0 = mblk * 128 + wave * 16;
    const int n0 = nblk * 64;
    const bf16_t* Arow = AO + (size_t)m0 * DM;
    const bf16_t* Wrow = Wb + (size_t)n0 * DM;

    const unsigned lds0 = lds_offset(&bpanel[0][0]);
    const unsigned lds1 = lds_offset(&bpanel[1][0]);
    const int NK = DM / 32;

    if (wave == 0) tdm_load_panel(lds0, Wrow, 64, DM);

    v8f acc[4] = {};
    int cur = 0;
    for (int k = 0; k < NK; ++k) {
        if (wave == 0) {
            if (k + 1 < NK) {
                tdm_load_panel(cur ? lds0 : lds1, Wrow + (k + 1) * 32, 64, DM);
                __builtin_amdgcn_s_wait_tensorcnt(1);
            } else {
                __builtin_amdgcn_s_wait_tensorcnt(0);
            }
        }
        __syncthreads();
        const v16bf a = load_a_frag(Arow, DM, k * 32, lane);
        const bf16_t* pan = &bpanel[cur][0];
        #pragma unroll
        for (int t = 0; t < 4; ++t)
            acc[t] = wmma_bf16(a, load_b_frag_lds(pan, t, lane), acc[t]);
        __syncthreads();
        cur ^= 1;
    }

    const int half = lane >> 4, col = lane & 15;
    #pragma unroll
    for (int t = 0; t < 4; ++t)
        #pragma unroll
        for (int j = 0; j < 8; ++j) {
            const int m = m0 + j + half * 8;
            out[(size_t)m * DM + n0 + t * 16 + col] = acc[t][j];
        }
}

// ---------------------------------------------------------------------------
// Host-side orchestration
// ---------------------------------------------------------------------------
extern "C" void kernel_launch(void* const* d_in, const int* in_sizes, int n_in,
                              void* d_out, int out_size, void* d_ws, size_t ws_size,
                              hipStream_t stream) {
    const float* x      = (const float*)d_in[0];   // [2,2048,1024]
    const int*   tokpos = (const int*)  d_in[1];   // [2048]
    const float* Wqkv   = (const float*)d_in[2];   // [3072,1024]
    const float* Wout   = (const float*)d_in[3];   // [1024,1024]
    float*       out    = (float*)d_out;           // [2,2048,1024]

    char* ws = (char*)d_ws;
    size_t off = 0;
    bf16_t* Xb    = (bf16_t*)(ws + off); off += (size_t)NB * SEQ * DM * 2;      // 8 MB
    bf16_t* Wqkvb = (bf16_t*)(ws + off); off += (size_t)3 * DM * DM * 2;        // 6 MB
    bf16_t* Woutb = (bf16_t*)(ws + off); off += (size_t)DM * DM * 2;            // 2 MB
    bf16_t* Qb    = (bf16_t*)(ws + off); off += (size_t)NB * NH * SEQ * DK * 2; // 8 MB
    bf16_t* Kb    = (bf16_t*)(ws + off); off += (size_t)NB * NH * SEQ * DK * 2; // 8 MB
    bf16_t* Vtb   = (bf16_t*)(ws + off); off += (size_t)NB * NH * DK * SEQ * 2; // 8 MB
    bf16_t* AOb   = (bf16_t*)(ws + off); off += (size_t)NB * SEQ * DM * 2;      // 8 MB

    const int nX = NB * SEQ * DM, nWq = 3 * DM * DM, nWo = DM * DM;
    cvt_f32_bf16<<<nX  / 256, 256, 0, stream>>>(x,    Xb,    nX);
    cvt_f32_bf16<<<nWq / 256, 256, 0, stream>>>(Wqkv, Wqkvb, nWq);
    cvt_f32_bf16<<<nWo / 256, 256, 0, stream>>>(Wout, Woutb, nWo);

    // (4096/128) * (3072/64) = 1536 blocks, 8 waves each
    qkv_rope_kernel<<<1536, 256, 0, stream>>>(Xb, Wqkvb, tokpos, Qb, Kb, Vtb);

    // 2*16*(2048/16) = 4096 wave-tiles
    flash_attn_kernel<<<512, 256, 0, stream>>>(Qb, Kb, Vtb, AOb);

    // (4096/128) * (1024/64) = 512 blocks
    out_proj_kernel<<<512, 256, 0, stream>>>(AOb, Woutb, out);
}
